// PredictiveLSTM_83021717831823
// MI455X (gfx1250) — compile-verified, matrix operand order checked by reading, not run on hardware
//
#include <hip/hip_runtime.h>
#include <hip/hip_bf16.h>

// ---------------------------------------------------------------------------
// LayerNorm-LSTM (4 layers) for MI455X / gfx1250.
//   * bf16 WMMA (v_wmma_f32_16x16x32_bf16), fp32 accumulate everywhere.
//   * Time-parallel input projections: one big WMMA GEMM per layer (Xprec).
//   * Sequential recurrence: ONE persistent kernel per layer (128 WGs, 1024
//     wave32s) looping over all 256 timesteps with software grid barriers —
//     eliminates ~2048 dependent kernel launches whose dispatch latency would
//     otherwise dominate (per-step GEMM is only 0.54 GFLOP / 8MB L2-resident
//     weight traffic).
// ---------------------------------------------------------------------------

#define LAYERS 4
#define ISZ   512
#define HSZ   1024
#define BSZ   64
#define TSZ   256
#define GSZ   (4 * HSZ)        // 4096
#define CATSZ (HSZ + ISZ)      // 1536
#define NWG_REC (GSZ / 32)     // 128 workgroups in the persistent kernel

typedef __attribute__((ext_vector_type(16))) __bf16       v16bf;
typedef __attribute__((ext_vector_type(8)))  float        v8f;
typedef __attribute__((ext_vector_type(4)))  unsigned int uivec4;

union FragU { uivec4 u[2]; v16bf v; };   // 32 bytes = 16 bf16

__device__ __forceinline__ unsigned short f2bf(float f) {
    unsigned u = __float_as_uint(f);
    u += 0x7FFFu + ((u >> 16) & 1u);     // round-to-nearest-even
    return (unsigned short)(u >> 16);
}
__device__ __forceinline__ float bf2f(unsigned short s) {
    return __uint_as_float(((unsigned)s) << 16);
}

// Sense-reversing grid barrier (agent scope). bar[0]=count, bar[1]=generation.
// Safe: 128 WGs x 8 waves co-reside trivially on an MI455X-class part.
__device__ __forceinline__ void grid_barrier(unsigned* bar, unsigned nwg) {
    __threadfence();             // release this thread's stores to device scope
    __syncthreads();
    if (threadIdx.x == 0) {
        unsigned g = __hip_atomic_load(&bar[1], __ATOMIC_RELAXED,
                                       __HIP_MEMORY_SCOPE_AGENT);
        unsigned a = __hip_atomic_fetch_add(&bar[0], 1u, __ATOMIC_ACQ_REL,
                                            __HIP_MEMORY_SCOPE_AGENT);
        if (a + 1 == nwg) {
            __hip_atomic_store(&bar[0], 0u, __ATOMIC_RELAXED,
                               __HIP_MEMORY_SCOPE_AGENT);
            __hip_atomic_fetch_add(&bar[1], 1u, __ATOMIC_RELEASE,
                                   __HIP_MEMORY_SCOPE_AGENT);
        } else {
            while (__hip_atomic_load(&bar[1], __ATOMIC_ACQUIRE,
                                     __HIP_MEMORY_SCOPE_AGENT) == g)
                __builtin_amdgcn_s_sleep(2);
        }
    }
    __syncthreads();
    __threadfence();             // acquire side
}

// ---------------------------------------------------------------------------
// Setup kernels
// ---------------------------------------------------------------------------
__global__ void k_cvt_bf16(const float* __restrict__ src,
                           unsigned short* __restrict__ dst, int n) {
    int i = blockIdx.x * blockDim.x + threadIdx.x;
    if (i < n) dst[i] = f2bf(src[i]);
}

// x[b][t][k] (B,T,I) -> inbuf[t*B+b][HSZ + k] (bf16 concat buffer, ld=CATSZ)
__global__ void k_pack_x(const float* __restrict__ x,
                         unsigned short* __restrict__ inbuf) {
    int i = blockIdx.x * blockDim.x + threadIdx.x;
    if (i >= BSZ * TSZ * ISZ) return;
    int k = i % ISZ;
    int t = (i / ISZ) % TSZ;
    int b = i / (ISZ * TSZ);
    inbuf[(size_t)(t * BSZ + b) * CATSZ + HSZ + k] = f2bf(x[i]);
}

// zero c-state (fp32), h bf16 buffer, and the grid-barrier words
__global__ void k_zero_state(float* __restrict__ cs,
                             unsigned short* __restrict__ hb,
                             unsigned* __restrict__ bar) {
    int i = blockIdx.x * blockDim.x + threadIdx.x;
    if (i < BSZ * HSZ) { cs[i] = 0.0f; hb[i] = 0; }
    if (i < 2) bar[i] = 0u;
}

// ---------------------------------------------------------------------------
// Big time-parallel GEMM: Xprec[m, g] = sum_k A[m,k] * W[g,k] + bias[g]
//   A: [TSZ*BSZ, lda] bf16 (row m = t*B+b), W: [GSZ, ldw] bf16, out bf16.
// Block = 256 thr = 8 waves; block tile 64(M) x 32(N); wave tile 16x16.
// Grid  = (GSZ/32, TSZ*BSZ/64).
// Fragment layout per ISA 7.12.2 (wave32): lane = kh*16 + m; per 32-K step a
// lane loads two 16B segments at k + kh*8 and k + 16 + kh*8.
// ---------------------------------------------------------------------------
__global__ void k_gemm_xprec(const unsigned short* __restrict__ A, int lda, int K,
                             const unsigned short* __restrict__ W, int ldw,
                             const float* __restrict__ bias,
                             unsigned short* __restrict__ Xp) {
    const int lane = threadIdx.x & 31;
    const int wave = threadIdx.x >> 5;
    const int n0 = blockIdx.x * 32 + (wave >> 2) * 16;
    const int m0 = blockIdx.y * 64 + (wave & 3) * 16;
    const int mn = lane & 15;
    const int kh = (lane >> 4) & 1;

    const unsigned short* rowA = A + (size_t)(m0 + mn) * lda + kh * 8;
    const unsigned short* rowB = W + (size_t)(n0 + mn) * ldw + kh * 8;

    v8f acc = {};
    for (int k = 0; k < K; k += 32) {
        FragU fa, fb;
        fa.u[0] = *(const uivec4*)(rowA + k);
        fa.u[1] = *(const uivec4*)(rowA + k + 16);
        fb.u[0] = *(const uivec4*)(rowB + k);
        fb.u[1] = *(const uivec4*)(rowB + k + 16);
        acc = __builtin_amdgcn_wmma_f32_16x16x32_bf16(
                  false, fa.v, false, fb.v, (short)0, acc, false, false);
    }

    const int n = n0 + mn;
    const float bv = bias[n];
#pragma unroll
    for (int r = 0; r < 8; ++r) {                 // C/D: M = r + 8*kh
        int m = m0 + r + 8 * kh;
        Xp[(size_t)m * GSZ + n] = f2bf(acc[r] + bv);
    }
}

// ---------------------------------------------------------------------------
// Persistent per-layer recurrence. Grid = 128 WGs x 256 threads (1024 waves).
// Per timestep:
//   Phase A: gates[m,g] = Xprec[t*B+m,g] + h[m,:]·Wh[g,:]  (WMMA, 64x4096)
//   grid barrier
//   Phase B: WGs 0..63 -> joint LayerNorm over 4096 gates + LSTM cell for
//            batch b = blockIdx.x; h written as bf16 for next step / layer.
//   grid barrier
// ---------------------------------------------------------------------------
__global__ void k_layer_recurrent(const unsigned short* __restrict__ Whl,
                                  const unsigned short* __restrict__ Xp,
                                  const float* __restrict__ gamma,
                                  const float* __restrict__ beta,
                                  float* __restrict__ gates,
                                  float* __restrict__ cs,
                                  unsigned short* __restrict__ hb,
                                  unsigned short* __restrict__ inbuf,
                                  float* __restrict__ out,
                                  unsigned* __restrict__ bar,
                                  int l) {
    const int lane = threadIdx.x & 31;
    const int wave = threadIdx.x >> 5;
    const int mn = lane & 15;
    const int kh = (lane >> 4) & 1;
    const int n0 = blockIdx.x * 32 + (wave >> 2) * 16;
    const int m0 = (wave & 3) * 16;
    const int n  = n0 + mn;
    const unsigned short* rowA = hb  + (size_t)(m0 + mn) * HSZ + kh * 8;
    const unsigned short* rowB = Whl + (size_t)(n0 + mn) * HSZ + kh * 8;

    __shared__ float red[2][8];

    for (int t = 0; t < TSZ; ++t) {
        // ---------------- Phase A: WMMA gates GEMM -------------------------
        const unsigned short* Xpt = Xp + (size_t)t * BSZ * GSZ;
        v8f acc;
#pragma unroll
        for (int r = 0; r < 8; ++r)
            acc[r] = bf2f(Xpt[(size_t)(m0 + r + 8 * kh) * GSZ + n]);

        for (int k = 0; k < HSZ; k += 32) {
            __builtin_prefetch(rowB + k + 512, 0, 1);   // global_prefetch_b8
            FragU fa, fb;
            fa.u[0] = *(const uivec4*)(rowA + k);
            fa.u[1] = *(const uivec4*)(rowA + k + 16);
            fb.u[0] = *(const uivec4*)(rowB + k);
            fb.u[1] = *(const uivec4*)(rowB + k + 16);
            acc = __builtin_amdgcn_wmma_f32_16x16x32_bf16(
                      false, fa.v, false, fb.v, (short)0, acc, false, false);
        }
#pragma unroll
        for (int r = 0; r < 8; ++r)
            gates[(size_t)(m0 + r + 8 * kh) * GSZ + n] = acc[r];

        grid_barrier(bar, NWG_REC);

        // ---------------- Phase B: joint LN + LSTM cell --------------------
        if (blockIdx.x < BSZ) {
            const int b = blockIdx.x;
            const int tid = threadIdx.x;
            const float* grow = gates + (size_t)b * GSZ;

            float v[16];
            float s = 0.0f, s2 = 0.0f;
#pragma unroll
            for (int q = 0; q < 4; ++q) {
                int j = tid + 256 * q;
#pragma unroll
                for (int p = 0; p < 4; ++p) {
                    float x = grow[j + p * HSZ];
                    v[q * 4 + p] = x;
                    s += x; s2 += x * x;
                }
            }
            for (int off = 16; off > 0; off >>= 1) {
                s  += __shfl_down(s,  off, 32);
                s2 += __shfl_down(s2, off, 32);
            }
            const int w = tid >> 5;
            if (lane == 0) { red[0][w] = s; red[1][w] = s2; }
            __syncthreads();
            if (w == 0) {
                s  = (lane < 8) ? red[0][lane] : 0.0f;
                s2 = (lane < 8) ? red[1][lane] : 0.0f;
                for (int off = 4; off > 0; off >>= 1) {
                    s  += __shfl_down(s,  off, 32);
                    s2 += __shfl_down(s2, off, 32);
                }
                if (lane == 0) { red[0][0] = s; red[1][0] = s2; }
            }
            __syncthreads();
            const float mean = red[0][0] * (1.0f / GSZ);
            const float var  = red[1][0] * (1.0f / GSZ) - mean * mean;
            const float rstd = rsqrtf(var + 1e-5f);

#pragma unroll
            for (int q = 0; q < 4; ++q) {
                int j = tid + 256 * q;
                float gi = (v[q*4+0] - mean) * rstd * gamma[j          ] + beta[j          ];
                float gf = (v[q*4+1] - mean) * rstd * gamma[j +     HSZ] + beta[j +     HSZ];
                float gg = (v[q*4+2] - mean) * rstd * gamma[j + 2 * HSZ] + beta[j + 2 * HSZ];
                float go = (v[q*4+3] - mean) * rstd * gamma[j + 3 * HSZ] + beta[j + 3 * HSZ];
                float c = cs[(size_t)b * HSZ + j];
                float si = 1.0f / (1.0f + __expf(-gi));
                float sf = 1.0f / (1.0f + __expf(-gf));
                float so = 1.0f / (1.0f + __expf(-go));
                c = sf * c + si * tanhf(gg);
                float h = so * tanhf(c);
                cs[(size_t)b * HSZ + j] = c;
                unsigned short hbf = f2bf(h);
                hb[(size_t)b * HSZ + j] = hbf;                           // next step
                inbuf[(size_t)(t * BSZ + b) * CATSZ + j] = hbf;          // next layer
                out[((size_t)(t * HSZ + j) * BSZ + b) * LAYERS + l] = h; // output
            }
            __syncthreads();   // protect LDS 'red' before next t (b<64 WGs)
        }

        grid_barrier(bar, NWG_REC);
    }
}

// ---------------------------------------------------------------------------
extern "C" void kernel_launch(void* const* d_in, const int* in_sizes, int n_in,
                              void* d_out, int out_size, void* d_ws, size_t ws_size,
                              hipStream_t stream) {
    (void)in_sizes; (void)n_in; (void)out_size; (void)ws_size;
    const float* x   = (const float*)d_in[0];
    const float* Wx0 = (const float*)d_in[1];
    const float* Wh0 = (const float*)d_in[2];
    const float* b0  = (const float*)d_in[3];
    const float* g0  = (const float*)d_in[4];
    const float* be0 = (const float*)d_in[5];
    const float* Wx  = (const float*)d_in[6];
    const float* Wh  = (const float*)d_in[7];
    const float* bb  = (const float*)d_in[8];
    const float* gg  = (const float*)d_in[9];
    const float* be  = (const float*)d_in[10];
    float* out = (float*)d_out;

    char* ws = (char*)d_ws;
    size_t off = 0;
    auto carve = [&](size_t bytes) -> char* {
        char* p = ws + off;
        off += (bytes + 255) & ~(size_t)255;
        return p;
    };
    unsigned short* inbuf = (unsigned short*)carve((size_t)TSZ * BSZ * CATSZ * 2); //  50 MB
    unsigned short* Xp    = (unsigned short*)carve((size_t)TSZ * BSZ * GSZ * 2);   // 134 MB
    unsigned short* Wxb   = (unsigned short*)carve(((size_t)GSZ * ISZ + 3ull * GSZ * CATSZ) * 2);
    unsigned short* Whb   = (unsigned short*)carve((size_t)LAYERS * GSZ * HSZ * 2);
    float*          gates = (float*)carve((size_t)BSZ * GSZ * 4);
    float*          cs    = (float*)carve((size_t)BSZ * HSZ * 4);
    unsigned short* hb    = (unsigned short*)carve((size_t)BSZ * HSZ * 2);
    unsigned*       bar   = (unsigned*)carve(256);

    // One-time bf16 conversions + x packing
    int n;
    n = GSZ * ISZ;
    k_cvt_bf16<<<(n + 255) / 256, 256, 0, stream>>>(Wx0, Wxb, n);
    n = 3 * GSZ * CATSZ;
    k_cvt_bf16<<<(n + 255) / 256, 256, 0, stream>>>(Wx, Wxb + (size_t)GSZ * ISZ, n);
    n = GSZ * HSZ;
    k_cvt_bf16<<<(n + 255) / 256, 256, 0, stream>>>(Wh0, Whb, n);
    n = 3 * GSZ * HSZ;
    k_cvt_bf16<<<(n + 255) / 256, 256, 0, stream>>>(Wh, Whb + (size_t)GSZ * HSZ, n);
    n = BSZ * TSZ * ISZ;
    k_pack_x<<<(n + 255) / 256, 256, 0, stream>>>(x, inbuf);

    for (int l = 0; l < LAYERS; ++l) {
        const unsigned short* A;
        const unsigned short* Wp;
        const float *bias, *gamma, *beta;
        int K, ldw;
        if (l == 0) {
            A = inbuf + HSZ; K = ISZ;   Wp = Wxb; ldw = ISZ;
            bias = b0; gamma = g0; beta = be0;
        } else {
            A = inbuf;       K = CATSZ;
            Wp = Wxb + (size_t)GSZ * ISZ + (size_t)(l - 1) * GSZ * CATSZ; ldw = CATSZ;
            bias  = bb + (size_t)(l - 1) * GSZ;
            gamma = gg + (size_t)(l - 1) * GSZ;
            beta  = be + (size_t)(l - 1) * GSZ;
        }

        // Time-parallel input projection for this layer (big WMMA GEMM)
        k_gemm_xprec<<<dim3(GSZ / 32, (TSZ * BSZ) / 64), 256, 0, stream>>>(
            A, CATSZ, K, Wp, ldw, bias, Xp);

        // Reset recurrent state + barrier words
        k_zero_state<<<(BSZ * HSZ + 255) / 256, 256, 0, stream>>>(cs, hb, bar);

        // One persistent kernel runs the entire 256-step recurrence
        const unsigned short* Whl = Whb + (size_t)l * GSZ * HSZ;
        k_layer_recurrent<<<dim3(NWG_REC), 256, 0, stream>>>(
            Whl, Xp, gamma, beta, gates, cs, hb, inbuf, out, bar, l);
    }
}
